// LinearQ4_0_32023276159508
// MI455X (gfx1250) — compile-verified, matrix operand order checked by reading, not run on hardware
//
#include <hip/hip_runtime.h>

// ---- CDNA5 (gfx1250) Q4_0 linear: two-pass (dequant -> f16 WMMA GEMM) ----
// y[M,N] = x[M,K] @ deq(W,s)[N,K]^T,  M=8192, K=4096, N=11008, group=64.

typedef __attribute__((ext_vector_type(16))) _Float16 v16h;
typedef __attribute__((ext_vector_type(8)))  _Float16 v8h;
typedef __attribute__((ext_vector_type(8)))  float    v8f;
typedef __attribute__((ext_vector_type(8)))  signed char v8c;
typedef __attribute__((ext_vector_type(16))) signed char v16c;

#define K_DIM 4096
#define N_DIM 11008

union AFrag { v16h v; v8h h[2]; };
union BFrag { v16h v; _Float16 e[16]; };

// ---------------------------------------------------------------------------
// Pass 1: expand packed Q4_0 nibbles to f16 weights in workspace.
// deq flat element e = n*K + k; group g = e>>6; packed row = g>>1;
// g even -> arithmetic high nibble, g odd -> sign-extended low nibble.
// ---------------------------------------------------------------------------
__global__ __launch_bounds__(256)
void q40_dequant(const signed char* __restrict__ W,
                 const _Float16* __restrict__ S,
                 _Float16* __restrict__ Wf)
{
    const unsigned t  = blockIdx.x * blockDim.x + threadIdx.x;
    const unsigned e0 = t * 16u;               // first element of this chunk
    const unsigned g  = e0 >> 6;               // quant group (64 elems)
    const int      sel = (int)(g & 1u);        // 0: high nibble, 1: low nibble
    const unsigned p0  = e0 & 63u;             // position within group

    const _Float16 scale = S[g];
    const v16c raw = *(const v16c*)(W + (size_t)(g >> 1) * 64 + p0); // 16B aligned

    union { v8h v[2]; _Float16 e[16]; } out;
#pragma unroll
    for (int i = 0; i < 16; ++i) {
        signed char b = raw[i];
        int q = sel ? (int)((signed char)(b << 4) >> 4)   // sign-extend low 4
                    : (int)(b >> 4);                      // arithmetic high 4
        out.e[i] = (_Float16)q * scale;
    }
    v8h* dst = (v8h*)(Wf + e0);
    dst[0] = out.v[0];
    dst[1] = out.v[1];
}

// ---------------------------------------------------------------------------
// Pass 2: pure f16 WMMA GEMM. Wave tile 64M x 64N (4x4 16x16 accums),
// block 128M x 256N with 8 waves (2x4). K-step 32: 16 b128 loads, 16 WMMAs.
// Addresses kept as 32-bit element offsets off the kernel-arg bases so the
// backend emits global_load_b128 (SADDR+voffset), not flat_load.
// ---------------------------------------------------------------------------
__global__ __launch_bounds__(256)
void f16_wmma_gemm(const _Float16* __restrict__ X,
                   const _Float16* __restrict__ Wf,
                   _Float16* __restrict__ Y)
{
    const int lane  = threadIdx.x & 31;
    const int wave  = threadIdx.x >> 5;
    const int waveM = wave & 1;    // 2 waves along M
    const int waveN = wave >> 1;   // 4 waves along N

    const int m0 = blockIdx.x * 128 + waveM * 64;
    const int n0 = blockIdx.y * 256 + waveN * 64;

    const int l15    = lane & 15;
    const int half16 = lane >> 4;

    v8f acc[4][4] = {};

    unsigned offA[4], offB[4];   // element offsets (fit in 32 bits: <64M elems)
#pragma unroll
    for (int i = 0; i < 4; ++i)
        offA[i] = (unsigned)(m0 + i * 16 + l15) * (unsigned)K_DIM
                + (unsigned)(half16 * 8);
#pragma unroll
    for (int j = 0; j < 4; ++j)
        offB[j] = (unsigned)(n0 + j * 16 + l15) * (unsigned)K_DIM
                + (unsigned)(half16 * 8);

    for (unsigned k0 = 0; k0 < K_DIM; k0 += 32) {
        v16h a[4], b[4];
#pragma unroll
        for (int i = 0; i < 4; ++i) {
            AFrag u;
            u.h[0] = *(const v8h*)(X + offA[i] + k0);        // K = k0+h*8 ..+7
            u.h[1] = *(const v8h*)(X + offA[i] + k0 + 16);   // K = k0+16+h*8 ..+7
            a[i] = u.v;
        }
#pragma unroll
        for (int j = 0; j < 4; ++j) {
            AFrag u;
            u.h[0] = *(const v8h*)(Wf + offB[j] + k0);
            u.h[1] = *(const v8h*)(Wf + offB[j] + k0 + 16);
            b[j] = u.v;
        }
#pragma unroll
        for (int i = 0; i < 4; ++i)
#pragma unroll
            for (int j = 0; j < 4; ++j)
                acc[i][j] = __builtin_amdgcn_wmma_f32_16x16x32_f16(
                    false, a[i], false, b[j], (short)0, acc[i][j], false, false);
    }

#pragma unroll
    for (int i = 0; i < 4; ++i) {
#pragma unroll
        for (int j = 0; j < 4; ++j) {
            const int n = n0 + j * 16 + l15;
#pragma unroll
            for (int v = 0; v < 8; ++v) {
                const int m = m0 + i * 16 + half16 * 8 + v;
                Y[(size_t)m * N_DIM + n] = (_Float16)acc[i][j][v];
            }
        }
    }
}

// ---------------------------------------------------------------------------
// Fallback: fused dequant-in-loop kernel (used only if workspace is small).
// ---------------------------------------------------------------------------
__global__ __launch_bounds__(256)
void q40_wmma_gemm_fused(const _Float16* __restrict__ X,
                         const signed char* __restrict__ W,
                         const _Float16* __restrict__ S,
                         _Float16* __restrict__ Y)
{
    const int lane  = threadIdx.x & 31;
    const int wave  = threadIdx.x >> 5;
    const int waveM = wave & 1;
    const int waveN = wave >> 1;

    const int m0 = blockIdx.x * 128 + waveM * 64;
    const int n0 = blockIdx.y * 128 + waveN * 32;

    const int l15    = lane & 15;
    const int half16 = lane >> 4;

    v8f acc[4][2] = {};

    for (int k0 = 0; k0 < K_DIM; k0 += 32) {
        v16h a[4];
#pragma unroll
        for (int i = 0; i < 4; ++i) {
            const _Float16* p =
                X + (size_t)(m0 + i * 16 + l15) * K_DIM + k0 + half16 * 8;
            AFrag u;
            u.h[0] = *(const v8h*)(p);
            u.h[1] = *(const v8h*)(p + 16);
            a[i] = u.v;
        }
        const int j64 = k0 >> 6;
        const int sel = j64 & 1;
        v16h b[2];
#pragma unroll
        for (int j = 0; j < 2; ++j) {
            const int n = n0 + j * 16 + l15;
            const size_t r = (size_t)n * 32 + (k0 >> 7);
            const int pbase = (k0 & 63) + half16 * 8;
            const signed char* wp = W + r * 64 + pbase;
            const _Float16 scale = S[(size_t)n * 64 + j64];

            v8c raw0 = *(const v8c*)(wp);
            v8c raw1 = *(const v8c*)(wp + 16);

            BFrag ub;
#pragma unroll
            for (int tt = 0; tt < 8; ++tt) {
                signed char b0 = raw0[tt];
                signed char b1 = raw1[tt];
                int q0, q1;
                if (sel) {
                    q0 = (int)((signed char)(b0 << 4) >> 4);
                    q1 = (int)((signed char)(b1 << 4) >> 4);
                } else {
                    q0 = (int)(b0 >> 4);
                    q1 = (int)(b1 >> 4);
                }
                ub.e[tt]     = (_Float16)q0 * scale;
                ub.e[8 + tt] = (_Float16)q1 * scale;
            }
            b[j] = ub.v;
        }
#pragma unroll
        for (int i = 0; i < 4; ++i)
#pragma unroll
            for (int j = 0; j < 2; ++j)
                acc[i][j] = __builtin_amdgcn_wmma_f32_16x16x32_f16(
                    false, a[i], false, b[j], (short)0, acc[i][j], false, false);
    }

#pragma unroll
    for (int i = 0; i < 4; ++i) {
#pragma unroll
        for (int j = 0; j < 2; ++j) {
            const int n = n0 + j * 16 + l15;
#pragma unroll
            for (int v = 0; v < 8; ++v) {
                const int m = m0 + i * 16 + half16 * 8 + v;
                Y[(size_t)m * N_DIM + n] = (_Float16)acc[i][j][v];
            }
        }
    }
}

extern "C" void kernel_launch(void* const* d_in, const int* in_sizes, int n_in,
                              void* d_out, int out_size, void* d_ws, size_t ws_size,
                              hipStream_t stream) {
    const _Float16*    X = (const _Float16*)d_in[0];     // (4,2048,4096) f16
    const signed char* W = (const signed char*)d_in[1];  // packed nibbles
    const _Float16*    S = (const _Float16*)d_in[2];     // f16 scales
    _Float16*          Y = (_Float16*)d_out;             // (4,2048,11008) f16

    const int M = in_sizes[0] / K_DIM;                   // 8192
    const size_t wf_bytes = (size_t)N_DIM * K_DIM * sizeof(_Float16); // ~90 MB

    if (ws_size >= wf_bytes && (M % 128) == 0) {
        _Float16* Wf = (_Float16*)d_ws;
        // Pass 1: dequant (N*K/16 threads, 256/block)
        const unsigned nblocks = (unsigned)((size_t)N_DIM * K_DIM / 16 / 256);
        q40_dequant<<<dim3(nblocks), dim3(256), 0, stream>>>(W, S, Wf);
        // Pass 2: f16 WMMA GEMM, block 128M x 256N
        dim3 grid(M / 128, N_DIM / 256);                 // (64, 43)
        f16_wmma_gemm<<<grid, dim3(256), 0, stream>>>(X, Wf, Y);
    } else {
        dim3 grid(M / 128, N_DIM / 128);                 // (64, 86)
        q40_wmma_gemm_fused<<<grid, dim3(256), 0, stream>>>(X, W, S, Y);
    }
}
